// MultiHeadAttention_10385230922057
// MI455X (gfx1250) — compile-verified, matrix operand order checked by reading, not run on hardware
//
#include <hip/hip_runtime.h>
#include <hip/hip_bf16.h>

// ---- problem constants ----
#define Bc 2
#define Sq 2048
#define Ec 1024
#define Hc 16
#define Dc 64
#define Mrows (Bc * Sq)          // 4096
#define SCALE_Q 0.125f           // 64^-0.5

typedef __attribute__((ext_vector_type(16))) __bf16 v16bf;
typedef __attribute__((ext_vector_type(8)))  __bf16 v8bf;
typedef __attribute__((ext_vector_type(8)))  float  v8f;
typedef __attribute__((ext_vector_type(4)))  float  v4f;

// ---------------------------------------------------------------------------
// Fragment loaders. CDNA5 16-bit WMMA layout (ISA 7.12.2):
//   A (16x32, MxK): lane&15 = row; lanes 0-15 hold K = kb+[0..7], kb+[16..23];
//                   lanes 16-31 hold K = kb+[8..15], kb+[24..31].
//   B (32x16, KxN): symmetric — lane&15 = column N, same K striping.
// All four GEMMs are NT, so each lane's K-strip is two contiguous 16-byte
// runs: two global_load_b128 straight into the WMMA operand registers.
// ---------------------------------------------------------------------------
__device__ __forceinline__ v16bf frag_from_bf16(const __bf16* __restrict__ base,
                                                int ld, int r0, int kb) {
  const int lane = threadIdx.x & 31;
  const int row  = r0 + (lane & 15);
  const int koff = kb + ((lane >> 4) << 3);        // +0 or +8
  const __bf16* p = base + (size_t)row * ld + koff;
  v8bf lo = *(const v8bf*)(p);
  v8bf hi = *(const v8bf*)(p + 16);
  v16bf f;
#pragma unroll
  for (int i = 0; i < 8; ++i) { f[i] = lo[i]; f[8 + i] = hi[i]; }
  return f;
}

// f32 source, streamed once -> non-temporal loads (TH=NT), convert to bf16.
__device__ __forceinline__ v16bf frag_from_f32_nt(const float* __restrict__ base,
                                                  int ld, int r0, int kb) {
  const int lane = threadIdx.x & 31;
  const int row  = r0 + (lane & 15);
  const int koff = kb + ((lane >> 4) << 3);
  const float* p = base + (size_t)row * ld + koff;
  v4f x0 = __builtin_nontemporal_load((const v4f*)(p));
  v4f x1 = __builtin_nontemporal_load((const v4f*)(p + 4));
  v4f x2 = __builtin_nontemporal_load((const v4f*)(p + 16));
  v4f x3 = __builtin_nontemporal_load((const v4f*)(p + 20));
  v16bf f;
#pragma unroll
  for (int i = 0; i < 4; ++i) {
    f[i]      = (__bf16)x0[i];
    f[4 + i]  = (__bf16)x1[i];
    f[8 + i]  = (__bf16)x2[i];
    f[12 + i] = (__bf16)x3[i];
  }
  return f;
}

__device__ __forceinline__ v8f wmma_bf16(v16bf a, v16bf b, v8f c) {
  // D = A x B + C, f32 accumulate: v_wmma_f32_16x16x32_bf16
  return __builtin_amdgcn_wmma_f32_16x16x32_bf16(
      /*neg_a=*/false, a, /*neg_b=*/false, b,
      /*c_mod=*/(short)0, c, /*reuse_a=*/false, /*reuse_b=*/false);
}

// ---------------------------------------------------------------------------
// Kernel 0: one-shot f32 -> bf16 conversion (x and the four weight matrices).
// 8 elements per thread: 2x b128 loads, 1x b128 store.
// ---------------------------------------------------------------------------
__global__ void __launch_bounds__(256) cvt_f32_bf16_kernel(
    const float* __restrict__ src, __bf16* __restrict__ dst) {
  const int i = blockIdx.x * 256 + threadIdx.x;       // grid sized exactly
  const v4f* p = (const v4f*)src + 2 * (size_t)i;
  v4f a = p[0], b = p[1];
  v8bf o;
#pragma unroll
  for (int t = 0; t < 4; ++t) { o[t] = (__bf16)a[t]; o[4 + t] = (__bf16)b[t]; }
  *((v8bf*)dst + i) = o;
}

// ---------------------------------------------------------------------------
// Kernel 1: fused QKV projection.  y = x @ W^T + b for W in {Wq,Wk,Wv}.
// Wave tile: 16(M) x 32(N) x 3 outputs; A fragment reused 6x per k-step.
// Q stored pre-scaled; K as [b,h,s,d]; V stored transposed [b,h,d,s].
// waves = (4096/16) * (1024/32) = 8192 -> 1024 blocks x 8 waves
// ---------------------------------------------------------------------------
__global__ void __launch_bounds__(256) qkv_proj_kernel(
    const __bf16* __restrict__ Xb,
    const __bf16* __restrict__ Wqb, const float* __restrict__ bq,
    const __bf16* __restrict__ Wkb, const float* __restrict__ bk,
    const __bf16* __restrict__ Wvb, const float* __restrict__ bv,
    __bf16* __restrict__ Qb, __bf16* __restrict__ Kb, __bf16* __restrict__ Vt) {
  const int wave = blockIdx.x * 8 + (threadIdx.x >> 5);
  const int m0 = (wave >> 5) * 16;     // 256 m-tiles
  const int n0 = (wave & 31) * 32;     // 32 n-groups of 32
  v8f aq[2] = {}, ak[2] = {}, av[2] = {};
  for (int kb = 0; kb < Ec; kb += 32) {
    v16bf a = frag_from_bf16(Xb, Ec, m0, kb);
#pragma unroll
    for (int j = 0; j < 2; ++j) {
      v16bf b;
      b = frag_from_bf16(Wqb, Ec, n0 + 16 * j, kb); aq[j] = wmma_bf16(a, b, aq[j]);
      b = frag_from_bf16(Wkb, Ec, n0 + 16 * j, kb); ak[j] = wmma_bf16(a, b, ak[j]);
      b = frag_from_bf16(Wvb, Ec, n0 + 16 * j, kb); av[j] = wmma_bf16(a, b, av[j]);
    }
  }
  const int lane  = threadIdx.x & 31;
  const int col   = lane & 15;
  const int rbase = (lane >> 4) * 8;
#pragma unroll
  for (int j = 0; j < 2; ++j) {
    const int n = n0 + 16 * j + col;
    const int h = n >> 6, d = n & 63;
    const float biq = bq[n], bik = bk[n], biv = bv[n];
#pragma unroll
    for (int r = 0; r < 8; ++r) {
      const int m = m0 + r + rbase;
      const int b = m >> 11, s = m & 2047;
      const size_t qk = (((size_t)b * Hc + h) * Sq + s) * Dc + d;   // [b,h,s,d]
      Qb[qk] = (__bf16)((aq[j][r] + biq) * SCALE_Q);
      Kb[qk] = (__bf16)(ak[j][r] + bik);
      const size_t vi = (((size_t)b * Hc + h) * Dc + d) * Sq + s;   // [b,h,d,s]
      Vt[vi] = (__bf16)(av[j][r] + biv);
    }
  }
}

// ---------------------------------------------------------------------------
// Kernel 2: scores[bh] = Qb[bh] @ Kb[bh]^T  (scale already folded into Q).
// Wave tile: 16(M) x 128(N), K = 64 = 2 WMMA steps; A fragments reused 8x.
// Scores stream out exactly once -> non-temporal stores.
// waves = 32 * 128 * 16 = 65536 -> 8192 blocks
// ---------------------------------------------------------------------------
__global__ void __launch_bounds__(256) attn_scores_kernel(
    const __bf16* __restrict__ Qb, const __bf16* __restrict__ Kb,
    float* __restrict__ scores) {
  const int wave = blockIdx.x * 8 + (threadIdx.x >> 5);
  const int bh   = wave >> 11;
  const int rem  = wave & 2047;
  const int m0   = (rem >> 4) * 16;
  const int n0   = (rem & 15) * 128;
  const __bf16* qbase = Qb + (size_t)bh * Sq * Dc;
  const __bf16* kbase = Kb + (size_t)bh * Sq * Dc;
  float* out = scores + (size_t)bh * Sq * Sq;
  v16bf a0 = frag_from_bf16(qbase, Dc, m0, 0);
  v16bf a1 = frag_from_bf16(qbase, Dc, m0, 32);
  const int lane  = threadIdx.x & 31;
  const int col   = lane & 15;
  const int rbase = (lane >> 4) * 8;
#pragma unroll
  for (int j = 0; j < 8; ++j) {
    v16bf b0 = frag_from_bf16(kbase, Dc, n0 + 16 * j, 0);
    v16bf b1 = frag_from_bf16(kbase, Dc, n0 + 16 * j, 32);
    v8f acc = {};
    acc = wmma_bf16(a0, b0, acc);
    acc = wmma_bf16(a1, b1, acc);
#pragma unroll
    for (int r = 0; r < 8; ++r)
      __builtin_nontemporal_store(
          acc[r], &out[(size_t)(m0 + r + rbase) * Sq + (n0 + 16 * j + col)]);
  }
}

// ---------------------------------------------------------------------------
// Kernel 3: softmax over the HEADS axis (axis=1 of [B,H,S,S]) — faithful to
// the reference. In place, each element touched once -> NT load/store.
// ---------------------------------------------------------------------------
__global__ void __launch_bounds__(256) softmax_heads_kernel(float* __restrict__ attn) {
  const size_t idx = (size_t)blockIdx.x * 256 + threadIdx.x;   // B*S*S
  const size_t b   = idx >> 22;             // / (S*S)
  const size_t r   = idx & 4194303;
  const size_t stride = (size_t)Sq * Sq;
  float* p = attn + b * (size_t)Hc * stride + r;
  float vals[Hc];
  float mx = -3.4e38f;
#pragma unroll
  for (int h = 0; h < Hc; ++h) {
    vals[h] = __builtin_nontemporal_load(&p[(size_t)h * stride]);
    mx = fmaxf(mx, vals[h]);
  }
  float sum = 0.f;
#pragma unroll
  for (int h = 0; h < Hc; ++h) { vals[h] = __expf(vals[h] - mx); sum += vals[h]; }
  const float inv = 1.0f / sum;
#pragma unroll
  for (int h = 0; h < Hc; ++h)
    __builtin_nontemporal_store(vals[h] * inv, &p[(size_t)h * stride]);
}

// ---------------------------------------------------------------------------
// Kernel 4: oh[b,s,h*64+d] = attn[bh] @ V[bh].  A = attn (f32, streamed once
// -> NT loads, bf16 on the fly); B = Vt rows (d-major, contiguous along k,
// L2-resident). Wave tile: 16(M) x 64(N=full D), K = 2048 (64 steps).
// waves = 32 * 128 = 4096 -> 512 blocks
// ---------------------------------------------------------------------------
__global__ void __launch_bounds__(256) attn_v_kernel(
    const float* __restrict__ attn, const __bf16* __restrict__ Vt,
    __bf16* __restrict__ OHb) {
  const int wave = blockIdx.x * 8 + (threadIdx.x >> 5);
  const int bh   = wave >> 7;
  const int m0   = (wave & 127) * 16;
  const int b    = bh >> 4, h = bh & 15;
  const float*  abase = attn + (size_t)bh * Sq * Sq;
  const __bf16* vbase = Vt + (size_t)bh * Dc * Sq;
  v8f acc[4] = {};
  for (int kb = 0; kb < Sq; kb += 32) {
    v16bf a = frag_from_f32_nt(abase, Sq, m0, kb);
#pragma unroll
    for (int j = 0; j < 4; ++j) {
      v16bf bf = frag_from_bf16(vbase, Sq, 16 * j, kb);
      acc[j] = wmma_bf16(a, bf, acc[j]);
    }
  }
  const int lane  = threadIdx.x & 31;
  const int col   = lane & 15;
  const int rbase = (lane >> 4) * 8;
#pragma unroll
  for (int j = 0; j < 4; ++j) {
    const int d = 16 * j + col;
#pragma unroll
    for (int r = 0; r < 8; ++r) {
      const int s = m0 + r + rbase;
      OHb[((size_t)b * Sq + s) * Ec + h * Dc + d] = (__bf16)acc[j][r];
    }
  }
}

// ---------------------------------------------------------------------------
// Kernel 5: out = oh @ Wo^T + bo.  Wave tile 16 x 64, K = 1024.
// Final output written once -> NT stores.
// waves = 256 * 16 = 4096 -> 512 blocks
// ---------------------------------------------------------------------------
__global__ void __launch_bounds__(256) out_proj_kernel(
    const __bf16* __restrict__ OHb, const __bf16* __restrict__ Wob,
    const float* __restrict__ bo, float* __restrict__ out) {
  const int wave = blockIdx.x * 8 + (threadIdx.x >> 5);
  const int m0 = (wave >> 4) * 16;
  const int n0 = (wave & 15) * 64;
  v8f acc[4] = {};
  for (int kb = 0; kb < Ec; kb += 32) {
    v16bf a = frag_from_bf16(OHb, Ec, m0, kb);
#pragma unroll
    for (int j = 0; j < 4; ++j) {
      v16bf b = frag_from_bf16(Wob, Ec, n0 + 16 * j, kb);
      acc[j] = wmma_bf16(a, b, acc[j]);
    }
  }
  const int lane  = threadIdx.x & 31;
  const int col   = lane & 15;
  const int rbase = (lane >> 4) * 8;
#pragma unroll
  for (int j = 0; j < 4; ++j) {
    const int n = n0 + 16 * j + col;
    const float bias = bo[n];
#pragma unroll
    for (int r = 0; r < 8; ++r)
      __builtin_nontemporal_store(acc[j][r] + bias,
                                  &out[(size_t)(m0 + r + rbase) * Ec + n]);
  }
}

// ---------------------------------------------------------------------------
extern "C" void kernel_launch(void* const* d_in, const int* in_sizes, int n_in,
                              void* d_out, int out_size, void* d_ws, size_t ws_size,
                              hipStream_t stream) {
  const float* x  = (const float*)d_in[0];
  const float* Wq = (const float*)d_in[1];
  const float* bq = (const float*)d_in[2];
  const float* Wk = (const float*)d_in[3];
  const float* bk = (const float*)d_in[4];
  const float* Wv = (const float*)d_in[5];
  const float* bv = (const float*)d_in[6];
  const float* Wo = (const float*)d_in[7];
  const float* bo = (const float*)d_in[8];

  float* out  = (float*)d_out;                                  // [B,S,E]
  float* attn = out + (size_t)Bc * Sq * Ec;                     // [B,H,S,S]

  char* ws = (char*)d_ws;
  const size_t MB = (size_t)1 << 20;
  __bf16* Xb  = (__bf16*)(ws);               //  8 MB  [4096,1024]
  __bf16* Wqb = (__bf16*)(ws +  8 * MB);     //  2 MB  [1024,1024]
  __bf16* Wkb = (__bf16*)(ws + 10 * MB);     //  2 MB
  __bf16* Wvb = (__bf16*)(ws + 12 * MB);     //  2 MB
  __bf16* Wob = (__bf16*)(ws + 14 * MB);     //  2 MB
  __bf16* Qb  = (__bf16*)(ws + 16 * MB);     //  8 MB  [b,h,s,d]
  __bf16* Kb  = (__bf16*)(ws + 24 * MB);     //  8 MB  [b,h,s,d]
  __bf16* Vt  = (__bf16*)(ws + 32 * MB);     //  8 MB  [b,h,d,s]
  __bf16* OHb = (__bf16*)(ws + 40 * MB);     //  8 MB  [B,S,E]

  // one-shot bf16 conversions (8 elems/thread; grids divide exactly)
  cvt_f32_bf16_kernel<<<2048, 256, 0, stream>>>(x,  Xb);   // 4,194,304 elems
  cvt_f32_bf16_kernel<<< 512, 256, 0, stream>>>(Wq, Wqb);  // 1,048,576 elems
  cvt_f32_bf16_kernel<<< 512, 256, 0, stream>>>(Wk, Wkb);
  cvt_f32_bf16_kernel<<< 512, 256, 0, stream>>>(Wv, Wvb);
  cvt_f32_bf16_kernel<<< 512, 256, 0, stream>>>(Wo, Wob);

  qkv_proj_kernel<<<1024, 256, 0, stream>>>(Xb, Wqb, bq, Wkb, bk, Wvb, bv,
                                            Qb, Kb, Vt);
  attn_scores_kernel<<<8192, 256, 0, stream>>>(Qb, Kb, attn);
  softmax_heads_kernel<<<32768, 256, 0, stream>>>(attn);
  attn_v_kernel<<<512, 256, 0, stream>>>(attn, Vt, OHb);
  out_proj_kernel<<<512, 256, 0, stream>>>(OHb, Wob, bo, out);
}
